// MambaRecursiveStateAdjustmentV2_41944650613241
// MI455X (gfx1250) — matrix operand order repaired and checked by reading, not compiled, not run on hardware
//
#include <hip/hip_runtime.h>
#include <hip/hip_bf16.h>
#include <math.h>

// Problem constants (match reference)
#define BATCH 2
#define CH 64
#define HH 64
#define WW 64
#define LL (HH*WW)        // 4096
#define DI 128            // D_INNER
#define EE (2*DI)         // 256
#define NS 8              // D_STATE
#define RR 4              // DT_RANK
#define KD 4              // K_DIRS
#define XP (RR + 2*NS)    // 20
#define LENGTH 2

typedef __attribute__((ext_vector_type(16))) _Float16 v16h;
typedef __attribute__((ext_vector_type(8)))  float    v8f;

// ---------- WMMA lane/element index helpers (CDNA5 ISA 7.12.2 layouts) ----------
__device__ __forceinline__ int wmma_ka(int lane, int j) {
  // 16-bit A matrix 16x32: lanes 0-15 hold K {0..7,16..23}, lanes 16-31 hold K {8..15,24..31}
  return j + ((j >= 8) ? 8 : 0) + ((lane >= 16) ? 8 : 0);
}
__device__ __forceinline__ int wmma_kb(int lane, int j) {
  // 16-bit B matrix 32x16: lanes 0-15 hold K {0..7,16..23}, lanes 16-31 hold K {8..15,24..31}; N = lane&15
  return j + ((j >= 8) ? 8 : 0) + ((lane >= 16) ? 8 : 0);
}

__device__ __forceinline__ float warp_sum(float v) {
  #pragma unroll
  for (int m = 16; m >= 1; m >>= 1) v += __shfl_xor(v, m, 32);
  return v;
}
__device__ __forceinline__ float warp_max(float v) {
  #pragma unroll
  for (int m = 16; m >= 1; m >>= 1) v = fmaxf(v, __shfl_xor(v, m, 32));
  return v;
}
__device__ __forceinline__ float siluf(float x) { return x / (1.f + expf(-x)); }
__device__ __forceinline__ float softplusf(float x) {
  return (x > 20.f) ? x : log1pf(expf(x));
}

// ---------- per-pixel layernorm over 64 channels (wave per pixel, 2 ch/lane) ----------
__global__ void ln_c64_kernel(const float* __restrict__ x, long long boff, long long bstride,
                              const float* __restrict__ g, const float* __restrict__ be,
                              float* __restrict__ out) {
  int lane = threadIdx.x & 31;
  int pix  = blockIdx.x * (blockDim.x >> 5) + (threadIdx.x >> 5);
  int b = pix / LL, l = pix - b * LL;
  const float* xp = x + boff + (long long)b * bstride + l;
  float v0 = xp[(long long)lane * LL];
  float v1 = xp[(long long)(lane + 32) * LL];
  float s  = warp_sum(v0 + v1);
  float sq = warp_sum(v0 * v0 + v1 * v1);
  float mu  = s * (1.f / CH);
  float var = sq * (1.f / CH) - mu * mu;
  float inv = rsqrtf(var + 1e-5f);
  float* op = out + ((long long)b * CH) * LL + l;
  op[(long long)lane * LL]        = (v0 - mu) * inv * g[lane] + be[lane];
  op[(long long)(lane + 32) * LL] = (v1 - mu) * inv * g[lane + 32] + be[lane + 32];
}

// ---------- conv_init (full 3x3, 64->64) fused with cur = prev * exp(conv) ----------
__global__ void conv_init_cur_kernel(const float* __restrict__ sig,
                                     const float* __restrict__ w,    // [CH][CH][3][3]
                                     const float* __restrict__ bias, // [CH]
                                     const float* __restrict__ prev,
                                     float* __restrict__ cur) {
  int t = blockIdx.x * blockDim.x + threadIdx.x;  // (b*CH+o)*LL + l
  int l = t % LL;
  int o = (t / LL) % CH;
  int b = t / (LL * CH);
  int y = l / WW, x = l % WW;
  float acc = bias[o];
  for (int i = 0; i < CH; ++i) {
    const float* sp = sig + (long long)(b * CH + i) * LL;
    const float* wp = w + (o * CH + i) * 9;
    #pragma unroll
    for (int ky = 0; ky < 3; ++ky) {
      int yy = y + ky - 1;
      if (yy < 0 || yy >= HH) continue;
      #pragma unroll
      for (int kx = 0; kx < 3; ++kx) {
        int xx = x + kx - 1;
        if (xx < 0 || xx >= WW) continue;
        acc += wp[ky * 3 + kx] * sp[yy * WW + xx];
      }
    }
  }
  cur[t] = prev[t] * expf(acc);
}

// ---------- KL divergence term per pixel over 64 channels ----------
__global__ void cal_kl_kernel(const float* __restrict__ cur, const float* __restrict__ prev,
                              float* __restrict__ out) {
  int lane = threadIdx.x & 31;
  int pix  = blockIdx.x * (blockDim.x >> 5) + (threadIdx.x >> 5);
  int b = pix / LL, l = pix - b * LL;
  long long base = (long long)b * CH * LL + l;
  float p0 = prev[base + (long long)lane * LL];
  float p1 = prev[base + (long long)(lane + 32) * LL];
  float c0 = cur[base + (long long)lane * LL];
  float c1 = cur[base + (long long)(lane + 32) * LL];
  float mp = warp_max(fmaxf(p0, p1));
  float mc = warp_max(fmaxf(c0, c1));
  float sp = warp_sum(expf(p0 - mp) + expf(p1 - mp));
  float sc = warp_sum(expf(c0 - mc) + expf(c1 - mc));
  float lsep = logf(sp) + mp;
  float lsec = logf(sc) + mc;
  float lp0 = p0 - lsep, lp1 = p1 - lsep;
  float lc0 = c0 - lsec, lc1 = c1 - lsec;
  out[base + (long long)lane * LL]        = expf(lp0) * (lp0 - lc0);
  out[base + (long long)(lane + 32) * LL] = expf(lp1) * (lp1 - lc1);
}

// ---------- in_proj: xz[b,e,l] = sum_c W[e,c]*x[b,c,l] + bias[e]  (WMMA, M=256,K=64,N=8192) ----------
__global__ void in_proj_wmma_kernel(const float* __restrict__ x,    // [B][CH][LL]
                                    const float* __restrict__ W,    // [EE][CH]
                                    const float* __restrict__ bias, // [EE]
                                    float* __restrict__ xz) {       // [B][EE][LL]
  const int lane = threadIdx.x & 31;
  const int wave = blockIdx.x * (blockDim.x >> 5) + (threadIdx.x >> 5);
  const int NT = (BATCH * LL) / 16;       // 512
  const int tile_m = wave / NT;           // 0..15
  const int tile_n = wave - tile_m * NT;
  const int col = lane & 15;
  const int p = tile_n * 16 + col;
  const int b = p / LL;
  const int l = p - b * LL;
  v8f acc = {};
  #pragma unroll
  for (int kc = 0; kc < 2; ++kc) {
    const int c0 = kc * 32;
    v16h a, bm;
    #pragma unroll
    for (int j = 0; j < 16; ++j) {
      int ca = c0 + wmma_ka(lane, j);
      a[j] = (_Float16)W[(tile_m * 16 + col) * CH + ca];
      int cb = c0 + wmma_kb(lane, j);
      bm[j] = (_Float16)x[(long long)(b * CH + cb) * LL + l];
    }
    acc = __builtin_amdgcn_wmma_f32_16x16x32_f16(false, a, false, bm, (short)0, acc, false, false);
  }
  const int row0 = tile_m * 16 + ((lane >= 16) ? 8 : 0);
  #pragma unroll
  for (int v = 0; v < 8; ++v) {
    int e = row0 + v;
    xz[(long long)(b * EE + e) * LL + l] = acc[v] + bias[e];
  }
}

// ---------- depthwise 3x3 conv + SiLU on first DI channels of xz ----------
__global__ void dwconv_silu_kernel(const float* __restrict__ xz,
                                   const float* __restrict__ w,    // [DI][1][3][3]
                                   const float* __restrict__ bias, // [DI]
                                   float* __restrict__ xc) {       // [B][DI][LL]
  int t = blockIdx.x * blockDim.x + threadIdx.x;  // (b*DI+d)*LL + l
  int l = t % LL;
  int d = (t / LL) % DI;
  int b = t / (LL * DI);
  int y = l / WW, x = l % WW;
  const float* sp = xz + (long long)(b * EE + d) * LL;
  const float* wp = w + d * 9;
  float acc = bias[d];
  #pragma unroll
  for (int ky = 0; ky < 3; ++ky) {
    int yy = y + ky - 1;
    if (yy < 0 || yy >= HH) continue;
    #pragma unroll
    for (int kx = 0; kx < 3; ++kx) {
      int xx = x + kx - 1;
      if (xx < 0 || xx >= WW) continue;
      acc += wp[ky * 3 + kx] * sp[yy * WW + xx];
    }
  }
  xc[t] = siluf(acc);
}

// ---------- cross_scan: 4 directional copies ----------
__global__ void cross_scan_kernel(const float* __restrict__ xc, float* __restrict__ xs) {
  int t = blockIdx.x * blockDim.x + threadIdx.x;  // ((b*KD+k)*DI+d)*LL + l
  int l = t % LL;
  int d = (t / LL) % DI;
  int k = (t / (LL * DI)) % KD;
  int b = t / (LL * DI * KD);
  int m = (k & 2) ? (LL - 1 - l) : l;
  int src = (k & 1) ? ((m % HH) * WW + m / HH) : m;
  xs[t] = xc[(long long)(b * DI + d) * LL + src];
}

// ---------- x_proj: xdbl[b,k,e,l] = sum_d W[k,e,d]*xs[b,k,d,l]  (WMMA, M=20->32,K=128) ----------
__global__ void x_proj_wmma_kernel(const float* __restrict__ xs,   // [B][KD][DI][LL]
                                   const float* __restrict__ W,    // [KD][XP][DI]
                                   float* __restrict__ xdbl) {     // [B][KD][XP][LL]
  const int lane = threadIdx.x & 31;
  const int wave = blockIdx.x * (blockDim.x >> 5) + (threadIdx.x >> 5);
  const int NT = LL / 16;  // 256
  const int bk = wave / (2 * NT);
  int rem = wave - bk * (2 * NT);
  const int tile_m = rem / NT;           // 0..1 (rows 0..31, rows>=20 zero)
  const int tile_n = rem - tile_m * NT;
  const int b = bk / KD;
  const int k = bk - b * KD;
  const int col = lane & 15;
  const int l = tile_n * 16 + col;
  v8f acc = {};
  #pragma unroll
  for (int kc = 0; kc < 4; ++kc) {
    const int d0 = kc * 32;
    v16h a, bm;
    #pragma unroll
    for (int j = 0; j < 16; ++j) {
      int da = d0 + wmma_ka(lane, j);
      int e = tile_m * 16 + col;
      a[j] = (e < XP) ? (_Float16)W[((long long)k * XP + e) * DI + da] : (_Float16)0.f;
      int db = d0 + wmma_kb(lane, j);
      bm[j] = (_Float16)xs[((long long)(b * KD + k) * DI + db) * LL + l];
    }
    acc = __builtin_amdgcn_wmma_f32_16x16x32_f16(false, a, false, bm, (short)0, acc, false, false);
  }
  const int row0 = tile_m * 16 + ((lane >= 16) ? 8 : 0);
  #pragma unroll
  for (int v = 0; v < 8; ++v) {
    int e = row0 + v;
    if (e < XP)
      xdbl[((long long)(b * KD + k) * XP + e) * LL + l] = acc[v];
  }
}

// ---------- selective scan, state-parallel: one thread per (b,k,d,n) chain ----------
// dt_proj + softplus fused inline (R=4). y reduced over the 8 states with
// wave32 shfl_xor within each aligned 8-lane group. Streams are prefetched
// 128 elements ahead once per 64-step chunk (global_prefetch_b8, speculative).
__global__ void scan_kernel(const float* __restrict__ xs,    // u [B][KD][DI][LL]
                            const float* __restrict__ xdbl,  // dt rows 0..3, Bm at RR, Cm at RR+NS
                            const float* __restrict__ A_log, // [KD][DI][NS]
                            const float* __restrict__ dt_w,  // [KD][DI][RR]
                            const float* __restrict__ dt_b,  // [KD][DI]
                            const float* __restrict__ Ds,    // [KD][DI]
                            float* __restrict__ ys) {        // [B][KD][DI][LL]
  int t = blockIdx.x * blockDim.x + threadIdx.x;  // 0..BATCH*KD*DI*NS-1 (8192)
  int n = t & (NS - 1);
  int g = t >> 3;                 // (b*KD+k)*DI + d
  int d = g % DI;
  int k = (g / DI) % KD;
  int b = g / (DI * KD);
  const long long kd = (long long)k * DI + d;
  float Ac = -expf(A_log[kd * NS + n]);
  float w0 = dt_w[kd * RR + 0], w1 = dt_w[kd * RR + 1];
  float w2 = dt_w[kd * RR + 2], w3 = dt_w[kd * RR + 3];
  float dbias = dt_b[kd];
  float Dv = Ds[kd];
  const long long bkd = ((long long)(b * KD + k) * DI + d) * LL;
  const float* up = xs + bkd;
  const float* xr = xdbl + (long long)(b * KD + k) * XP * LL;      // dt-rank rows
  const float* Bp = xr + (long long)(RR + n) * LL;
  const float* Cp = Bp + (long long)NS * LL;
  float* yp = ys + bkd;
  float h = 0.f;
  for (int l0 = 0; l0 < LL; l0 += 64) {
    // speculative prefetch two chunks (128 elements) ahead in each stream
    __builtin_prefetch(up + l0 + 128, 0, 0);
    __builtin_prefetch(Bp + l0 + 128, 0, 0);
    __builtin_prefetch(Cp + l0 + 128, 0, 0);
    __builtin_prefetch(xr + l0 + 128, 0, 0);
    for (int l = l0; l < l0 + 64; ++l) {
      float acc = dbias + w0 * xr[l] + w1 * xr[LL + l]
                        + w2 * xr[2 * LL + l] + w3 * xr[3 * LL + l];
      float dtv = softplusf(acc);
      float uv  = up[l];
      h = h * expf(dtv * Ac) + dtv * Bp[l] * uv;
      float yv = h * Cp[l];
      yv += __shfl_xor(yv, 1, 32);
      yv += __shfl_xor(yv, 2, 32);
      yv += __shfl_xor(yv, 4, 32);
      if (n == 0) yp[l] = yv + uv * Dv;
    }
  }
}

// ---------- cross_merge ----------
__global__ void merge_kernel(const float* __restrict__ ys, float* __restrict__ ym) {
  int t = blockIdx.x * blockDim.x + threadIdx.x;  // (b*DI+d)*LL + l
  int l = t % LL;
  int d = (t / LL) % DI;
  int b = t / (LL * DI);
  int yy = l / WW, xx = l % WW;
  int lT = xx * HH + yy;
  long long base = (long long)(b * KD) * DI * LL + (long long)d * LL;
  long long stride = (long long)DI * LL;
  ym[t] = ys[base + l] + ys[base + stride + lT]
        + ys[base + 2 * stride + (LL - 1 - l)] + ys[base + 3 * stride + (LL - 1 - lT)];
}

// ---------- layernorm over 128 channels + SiLU(z) gate, in-place ----------
__global__ void ln_d128_gate_kernel(float* __restrict__ y,          // [B][DI][LL]
                                    const float* __restrict__ g,
                                    const float* __restrict__ be,
                                    const float* __restrict__ xz) { // z at channel DI..
  int lane = threadIdx.x & 31;
  int pix  = blockIdx.x * (blockDim.x >> 5) + (threadIdx.x >> 5);
  int b = pix / LL, l = pix - b * LL;
  float* yp = y + (long long)b * DI * LL + l;
  const float* zp = xz + (long long)(b * EE + DI) * LL + l;
  float v[4];
  float s = 0.f, sq = 0.f;
  #pragma unroll
  for (int i = 0; i < 4; ++i) {
    v[i] = yp[(long long)(lane + 32 * i) * LL];
    s += v[i]; sq += v[i] * v[i];
  }
  s = warp_sum(s); sq = warp_sum(sq);
  float mu  = s * (1.f / DI);
  float var = sq * (1.f / DI) - mu * mu;
  float inv = rsqrtf(var + 1e-5f);
  #pragma unroll
  for (int i = 0; i < 4; ++i) {
    int d = lane + 32 * i;
    float z = zp[(long long)d * LL];
    yp[(long long)d * LL] = ((v[i] - mu) * inv * g[d] + be[d]) * siluf(z);
  }
}

// ---------- out_proj: out[b,c,l] = sum_d W[c,d]*y[b,d,l] + bias[c]  (WMMA, M=64,K=128) ----------
__global__ void out_proj_wmma_kernel(const float* __restrict__ y,    // [B][DI][LL]
                                     const float* __restrict__ W,    // [CH][DI]
                                     const float* __restrict__ bias, // [CH]
                                     float* __restrict__ out) {      // [B][CH][LL]
  const int lane = threadIdx.x & 31;
  const int wave = blockIdx.x * (blockDim.x >> 5) + (threadIdx.x >> 5);
  const int NT = (BATCH * LL) / 16;  // 512
  const int tile_m = wave / NT;      // 0..3
  const int tile_n = wave - tile_m * NT;
  const int col = lane & 15;
  const int p = tile_n * 16 + col;
  const int b = p / LL;
  const int l = p - b * LL;
  v8f acc = {};
  #pragma unroll
  for (int kc = 0; kc < 4; ++kc) {
    const int d0 = kc * 32;
    v16h a, bm;
    #pragma unroll
    for (int j = 0; j < 16; ++j) {
      int da = d0 + wmma_ka(lane, j);
      a[j] = (_Float16)W[(tile_m * 16 + col) * DI + da];
      int db = d0 + wmma_kb(lane, j);
      bm[j] = (_Float16)y[(long long)(b * DI + db) * LL + l];
    }
    acc = __builtin_amdgcn_wmma_f32_16x16x32_f16(false, a, false, bm, (short)0, acc, false, false);
  }
  const int row0 = tile_m * 16 + ((lane >= 16) ? 8 : 0);
  #pragma unroll
  for (int v = 0; v < 8; ++v) {
    int c = row0 + v;
    out[(long long)(b * CH + c) * LL + l] = acc[v] + bias[c];
  }
}

// ---------- nxt = cur * exp(ss + si) + bias ----------
__global__ void final_combine_kernel(const float* __restrict__ cur,
                                     const float* __restrict__ ss,
                                     const float* __restrict__ si,
                                     const float* __restrict__ bo,
                                     float* __restrict__ nxt) {
  int t = blockIdx.x * blockDim.x + threadIdx.x;
  nxt[t] = cur[t] * expf(ss[t] + si[t]) + bo[t];
}

// ======================= host side =======================

struct MambaParams {
  const float *in_proj_w, *in_proj_b, *conv_w, *conv_b, *x_proj_w,
              *dt_proj_w, *dt_proj_b, *A_log, *Ds,
              *out_norm_g, *out_norm_b, *out_proj_w, *out_proj_b;
};

static MambaParams get_mamba(void* const* d_in, int base) {
  MambaParams p;
  p.in_proj_w  = (const float*)d_in[base + 0];
  p.in_proj_b  = (const float*)d_in[base + 1];
  p.conv_w     = (const float*)d_in[base + 2];
  p.conv_b     = (const float*)d_in[base + 3];
  p.x_proj_w   = (const float*)d_in[base + 4];
  p.dt_proj_w  = (const float*)d_in[base + 5];
  p.dt_proj_b  = (const float*)d_in[base + 6];
  p.A_log      = (const float*)d_in[base + 7];
  p.Ds         = (const float*)d_in[base + 8];
  p.out_norm_g = (const float*)d_in[base + 9];
  p.out_norm_b = (const float*)d_in[base + 10];
  p.out_proj_w = (const float*)d_in[base + 11];
  p.out_proj_b = (const float*)d_in[base + 12];
  return p;
}

static void ss2d_run(const float* x, const MambaParams& p, float* out,
                     float* xz, float* xconv, float* xs, float* xdbl,
                     float* ysbuf, float* ymerge, hipStream_t stream) {
  in_proj_wmma_kernel <<<1024, 256, 0, stream>>>(x, p.in_proj_w, p.in_proj_b, xz);
  dwconv_silu_kernel  <<<4096, 256, 0, stream>>>(xz, p.conv_w, p.conv_b, xconv);
  cross_scan_kernel   <<<16384, 256, 0, stream>>>(xconv, xs);
  x_proj_wmma_kernel  <<<512, 256, 0, stream>>>(xs, p.x_proj_w, xdbl);
  scan_kernel         <<<32, 256, 0, stream>>>(xs, xdbl, p.A_log, p.dt_proj_w,
                                               p.dt_proj_b, p.Ds, ysbuf);
  merge_kernel        <<<4096, 256, 0, stream>>>(ysbuf, ymerge);
  ln_d128_gate_kernel <<<1024, 256, 0, stream>>>(ymerge, p.out_norm_g, p.out_norm_b, xz);
  out_proj_wmma_kernel<<<256, 256, 0, stream>>>(ymerge, p.out_proj_w, p.out_proj_b, out);
}

extern "C" void kernel_launch(void* const* d_in, const int* in_sizes, int n_in,
                              void* d_out, int out_size, void* d_ws, size_t ws_size,
                              hipStream_t stream) {
  (void)in_sizes; (void)n_in; (void)out_size; (void)ws_size;

  const float* image_seq   = (const float*)d_in[0];  // [B][LENGTH][CH][LL]
  const float* dz          = (const float*)d_in[1];  // [B][CH][LL]
  const float* sigma       = (const float*)d_in[2];  // [B][CH][LL]
  const float* norm_dz_g   = (const float*)d_in[3];
  const float* norm_dz_b   = (const float*)d_in[4];
  const float* norm_img_g  = (const float*)d_in[5];
  const float* norm_img_b  = (const float*)d_in[6];
  const float* conv_init_w = (const float*)d_in[7];
  const float* conv_init_b = (const float*)d_in[8];
  MambaParams m_image = get_mamba(d_in, 9);
  MambaParams m_state = get_mamba(d_in, 22);
  MambaParams m_bias  = get_mamba(d_in, 35);

  const size_t SZ_BCL  = (size_t)BATCH * CH * LL;       // 524288
  const size_t SZ_BEL  = (size_t)BATCH * EE * LL;       // 2097152
  const size_t SZ_BDL  = (size_t)BATCH * DI * LL;       // 1048576
  const size_t SZ_BKDL = (size_t)BATCH * KD * DI * LL;  // 4194304
  const size_t SZ_BKEL = (size_t)BATCH * KD * XP * LL;  // 655360

  float* w = (float*)d_ws;
  size_t off = 0;
  auto alloc = [&](size_t n) { float* p = w + off; off += n; return p; };
  float* prev      = alloc(SZ_BCL);
  float* cur       = alloc(SZ_BCL);
  float* nxt       = alloc(SZ_BCL);
  float* tmp       = alloc(SZ_BCL);
  float* sig_img   = alloc(SZ_BCL);
  float* sig_state = alloc(SZ_BCL);
  float* bias_out  = alloc(SZ_BCL);
  float* xz        = alloc(SZ_BEL);
  float* xconv     = alloc(SZ_BDL);
  float* xs        = alloc(SZ_BKDL);
  float* xdbl      = alloc(SZ_BKEL);
  float* ysbuf     = alloc(SZ_BKDL);
  float* ymerge    = alloc(SZ_BDL);

  // prev = layernorm_c(difficult_zone)
  ln_c64_kernel<<<1024, 256, 0, stream>>>(dz, 0LL, (long long)CH * LL,
                                          norm_dz_g, norm_dz_b, prev);
  // cur = prev * exp(conv_init(sigma))
  conv_init_cur_kernel<<<2048, 256, 0, stream>>>(sigma, conv_init_w, conv_init_b, prev, cur);

  for (int i = 0; i < LENGTH; ++i) {
    // sigma_image = ss2d(layernorm(image_i), m_image)
    ln_c64_kernel<<<1024, 256, 0, stream>>>(image_seq, (long long)i * CH * LL,
                                            (long long)LENGTH * CH * LL,
                                            norm_img_g, norm_img_b, tmp);
    ss2d_run(tmp, m_image, sig_img, xz, xconv, xs, xdbl, ysbuf, ymerge, stream);
    // sigma_state = ss2d(cal_kl(cur, prev), m_state)
    cal_kl_kernel<<<1024, 256, 0, stream>>>(cur, prev, tmp);
    ss2d_run(tmp, m_state, sig_state, xz, xconv, xs, xdbl, ysbuf, ymerge, stream);
    // bias = ss2d(cur, m_bias)
    ss2d_run(cur, m_bias, bias_out, xz, xconv, xs, xdbl, ysbuf, ymerge, stream);
    // nxt = cur * exp(sigma_state + sigma_image) + bias
    final_combine_kernel<<<2048, 256, 0, stream>>>(cur, sig_state, sig_img, bias_out, nxt);
    float* t = prev; prev = cur; cur = nxt; nxt = t;
  }

  hipMemcpyAsync(d_out, cur, SZ_BCL * sizeof(float), hipMemcpyDeviceToDevice, stream);
}
